// RecurrentUnit_47373489274933
// MI455X (gfx1250) — matrix operand order repaired and checked by reading, not compiled, not run on hardware
//
#include <hip/hip_runtime.h>
#include <hip/hip_bf16.h>
#include <math.h>

typedef __attribute__((ext_vector_type(16))) _Float16 v16h;
typedef __attribute__((ext_vector_type(8)))  _Float16 v8h;
typedef __attribute__((ext_vector_type(8)))  float    v8f;
typedef __attribute__((address_space(3)))    float    lds_f;

#define NPTS 8192
#define NB   2
#define HCH  128
#define KNN  16
#define NQ   8            // queries per workgroup in gru kernel
#define NPOS 128          // NQ * KNN
#define CINP 96           // padded varying input channels (67 -> 96)
#define XSTR 104          // Xs row stride (halves), 16B aligned
#define HSTR 136          // H-buffer row stride (halves), 16B aligned

// ---------------------------------------------------------------------------
// WMMA helpers: fragment layouts per CDNA5 ISA 7.12.2
// ---------------------------------------------------------------------------
__device__ __forceinline__ v8f zero8() {
  v8f z = {0.f,0.f,0.f,0.f,0.f,0.f,0.f,0.f};
  return z;
}

__device__ __forceinline__ v16h ldA(const _Float16* W, int kpad, int mt, int ks, int lane) {
  const _Float16* p = W + (size_t)(mt*16 + (lane & 15)) * kpad + ks*32 + ((lane >> 4) << 3);
  v8h lo = *(const v8h*)(p);
  v8h hi = *(const v8h*)(p + 16);
  v16h r;
#pragma unroll
  for (int i = 0; i < 8; ++i) { r[i] = lo[i]; r[i+8] = hi[i]; }
  return r;
}

__device__ __forceinline__ v16h ldB(const _Float16* X, int stride, int nt, int ks, int lane) {
  const _Float16* p = X + (size_t)(nt*16 + (lane & 15)) * stride + ks*32 + ((lane >> 4) << 3);
  v8h lo = *(const v8h*)(p);
  v8h hi = *(const v8h*)(p + 16);
  v16h r;
#pragma unroll
  for (int i = 0; i < 8; ++i) { r[i] = lo[i]; r[i+8] = hi[i]; }
  return r;
}

__device__ __forceinline__ v8f wmma_f16(v16h a, v16h b, v8f c) {
  return __builtin_amdgcn_wmma_f32_16x16x32_f16(false, a, false, b, (short)0, c, false, false);
}

template<int KS, typename EPI>
__device__ __forceinline__ void gemm8(const _Float16* W, int kpad,
                                      const _Float16* Xin, int xstr,
                                      int mt, int lane, EPI epi) {
  v16h a[KS];
#pragma unroll
  for (int ks = 0; ks < KS; ++ks) a[ks] = ldA(W, kpad, mt, ks, lane);
#pragma unroll
  for (int nt = 0; nt < 8; ++nt) {
    v8f acc = zero8();
#pragma unroll
    for (int ks = 0; ks < KS; ++ks)
      acc = wmma_f16(a[ks], ldB(Xin, xstr, nt, ks, lane), acc);
    epi(nt, acc);
  }
}

template<int KS, typename EPI>
__device__ __forceinline__ void gemm1(const _Float16* W, int kpad,
                                      const _Float16* Xin, int xstr,
                                      int mt, int lane, EPI epi) {
  v8f acc = zero8();
#pragma unroll
  for (int ks = 0; ks < KS; ++ks)
    acc = wmma_f16(ldA(W, kpad, mt, ks, lane), ldB(Xin, xstr, 0, ks, lane), acc);
  epi(acc);
}

__device__ __forceinline__ float leaky(float x) { return x > 0.f ? x : 0.1f * x; }
__device__ __forceinline__ float sigm(float x)  { return 1.f / (1.f + expf(-x)); }

// ---------------------------------------------------------------------------
// Weight conversion to f16 (padded layouts)
// ---------------------------------------------------------------------------
__global__ __launch_bounds__(256) void convert_kernel(
    const float* wr0, const float* wz0, const float* wh0,
    const float* wr12, const float* wz12, const float* wh12,
    const float* wfr, const float* wfz, const float* wfro,
    _Float16* w0f, _Float16* w12f, _Float16* wfusef, float* wcols)
{
  int tid = blockIdx.x * 256 + threadIdx.x;
  int stride = gridDim.x * 256;
  for (int i = tid; i < 3*HCH*CINP; i += stride) {
    int br = i / (HCH*CINP); int rem = i % (HCH*CINP);
    int ch = rem / CINP; int c = rem % CINP;
    const float* W = (br == 0) ? wr0 : (br == 1) ? wz0 : wh0;
    float v = (c < 67) ? W[ch*135 + c] : 0.0f;
    w0f[i] = (_Float16)v;
  }
  for (int i = tid; i < 3*2*HCH*HCH; i += stride) {
    int br = i / (2*HCH*HCH); int rem = i % (2*HCH*HCH);
    const float* W = (br == 0) ? wr12 : (br == 1) ? wz12 : wh12;
    w12f[i] = (_Float16)W[rem];
  }
  for (int i = tid; i < 3*HCH*HCH; i += stride) {
    int br = i / (HCH*HCH);
    const float* W = (br == 0) ? wfr : (br == 1) ? wfz : wfro;
    wfusef[i] = (_Float16)W[i % (HCH*HCH)];
  }
  for (int i = tid; i < 3*HCH*4; i += stride) {
    int br = i / (HCH*4); int rem = i % (HCH*4);
    int ch = rem >> 2; int c = rem & 3;
    const float* W = (br == 0) ? wr0 : (br == 1) ? wz0 : wh0;
    wcols[i] = W[ch*135 + 67 + c];
  }
}

// ---------------------------------------------------------------------------
// points2 -> f16, point-major [B][N][64] for vectorized gathers
// ---------------------------------------------------------------------------
__global__ __launch_bounds__(256) void p2half_kernel(const float* points2, _Float16* p2t)
{
  int i = blockIdx.x * 256 + threadIdx.x;     // over B*N*64, c fastest
  int c = i & 63;
  int n = (i >> 6) & (NPTS - 1);
  int b = i >> 19;
  p2t[i] = (_Float16)points2[((size_t)b*64 + c)*NPTS + n];
}

// ---------------------------------------------------------------------------
// prep0: cosine schedule scalars, time embedding MLP, te-weight contraction
// ---------------------------------------------------------------------------
__global__ __launch_bounds__(256) void prep0_kernel(
    const int* t_in, const float* w1, const float* b1,
    const float* w2, const float* b2,
    const float* wr0, const float* wz0, const float* wh0,
    float* sched, float* te_out, float* tconst)
{
  __shared__ float h1[NB][64];
  __shared__ float teS[NB][64];
  int tid = threadIdx.x;
  if (tid < NB) {
    int t = t_in[tid];
    const float s = 0.008f;
    float acp = 1.0f;
    for (int j = 0; j <= t; ++j) {
      float x0 = ((float)j * 1e-3f + s) / (1.0f + s) * 1.57079632679f;
      float x1 = ((float)(j+1) * 1e-3f + s) / (1.0f + s) * 1.57079632679f;
      float a0 = cosf(x0); a0 *= a0;
      float a1 = cosf(x1); a1 *= a1;
      float beta = 1.0f - a1 / a0;
      beta = fminf(fmaxf(beta, 0.0f), 0.999f);
      acp *= (1.0f - beta);
    }
    sched[tid*2+0] = sqrtf(acp);
    sched[tid*2+1] = sqrtf(fmaxf(0.f, 1.0f - acp));
  }
  if (tid < NB*64) {
    int bb = tid >> 6, j = tid & 63;
    float tval = (float)t_in[bb];
    float acc = b1[j];
#pragma unroll
    for (int i = 0; i < 8; ++i) {
      float f = expf((float)i * (-9.210340371976184f / 7.0f));
      float ang = tval * f;
      acc += sinf(ang) * w1[j*16 + i] + cosf(ang) * w1[j*16 + 8 + i];
    }
    h1[bb][j] = 0.5f * acc * (1.0f + erff(acc * 0.70710678118f));
  }
  __syncthreads();
  if (tid < NB*64) {
    int bb = tid >> 6, j = tid & 63;
    float acc = b2[j];
    for (int i = 0; i < 64; ++i) acc += h1[bb][i] * w2[j*64 + i];
    teS[bb][j] = acc;
    te_out[bb*64 + j] = acc;
  }
  __syncthreads();
  for (int i = tid; i < 3*NB*HCH; i += 256) {
    int br = i / (NB*HCH); int rem = i % (NB*HCH);
    int bb = rem >> 7, ch = rem & 127;
    const float* W = (br == 0) ? wr0 : (br == 1) ? wz0 : wh0;
    float acc = 0.f;
    for (int k2 = 0; k2 < 64; ++k2) acc += teS[bb][k2] * W[ch*135 + 71 + k2];
    tconst[(br*NB + bb)*HCH + ch] = acc;
  }
}

// ---------------------------------------------------------------------------
// prep1: diffusion deltas per (b,n)
// ---------------------------------------------------------------------------
__global__ __launch_bounds__(256) void prep1_kernel(
    const float* flow, const float* flow_gt, const float* cert,
    const float* noise, const float* noise_cert,
    const float* sched, float* dfbuf, float* fnbuf)
{
  int i = blockIdx.x * 256 + threadIdx.x;     // over B*N
  int b = i >> 13, n = i & (NPTS-1);
  float a  = sched[b*2+0];
  float bc = sched[b*2+1];
  float fg0 = flow_gt[(size_t)i*3+0], fg1 = flow_gt[(size_t)i*3+1], fg2 = flow_gt[(size_t)i*3+2];
  float f0 = flow[((size_t)b*3+0)*NPTS + n];
  float f1 = flow[((size_t)b*3+1)*NPTS + n];
  float f2 = flow[((size_t)b*3+2)*NPTS + n];
  float d0 = fg0 - f0, d1 = fg1 - f1, d2 = fg2 - f2;
  float gtn = sqrtf(d0*d0 + d1*d1 + d2*d2);
  float sfn = sqrtf(fg0*fg0 + fg1*fg1 + fg2*fg2);
  float rel = gtn / (sfn + 1e-4f);
  float gc = (gtn < 0.5f || rel < 0.5f) ? 1.0f : 0.0f;
  float c = cert[(size_t)b*NPTS + n];
  float dc = gc - c;
  if (__builtin_isinf(d0)) d0 = 0.f;
  if (__builtin_isinf(d1)) d1 = 0.f;
  if (__builtin_isinf(d2)) d2 = 0.f;
  float df0 = a*d0 + bc*noise[((size_t)b*3+0)*NPTS + n];
  float df1 = a*d1 + bc*noise[((size_t)b*3+1)*NPTS + n];
  float df2 = a*d2 + bc*noise[((size_t)b*3+2)*NPTS + n];
  float dce = a*dc + bc*noise_cert[i];
  dfbuf[(size_t)i*4+0] = df0; dfbuf[(size_t)i*4+1] = df1;
  dfbuf[(size_t)i*4+2] = df2; dfbuf[(size_t)i*4+3] = dce;
  fnbuf[(size_t)i*4+0] = f0 + df0; fnbuf[(size_t)i*4+1] = f1 + df1;
  fnbuf[(size_t)i*4+2] = f2 + df2; fnbuf[(size_t)i*4+3] = c + dce;
}

// ---------------------------------------------------------------------------
// KNN: brute force; candidate tiles staged via async global->LDS DMA
// ---------------------------------------------------------------------------
__global__ __launch_bounds__(256) void knn_kernel(const float* xyz1, const float* xyz2, int* idxb)
{
  __shared__ float cs[3][256];
  int b = blockIdx.x / (NPTS/256);
  int n = (blockIdx.x % (NPTS/256)) * 256 + threadIdx.x;
  const float* x1 = xyz1 + (size_t)b*3*NPTS;
  const float* x2 = xyz2 + (size_t)b*3*NPTS;
  float px = x1[n], py = x1[NPTS + n], pz = x1[2*NPTS + n];
  float bd[KNN]; int bi[KNN];
#pragma unroll
  for (int j = 0; j < KNN; ++j) { bd[j] = 3.0e38f; bi[j] = 0; }

  unsigned l0 = (unsigned)(uintptr_t)(lds_f*)&cs[0][threadIdx.x];
  unsigned l1 = (unsigned)(uintptr_t)(lds_f*)&cs[1][threadIdx.x];
  unsigned l2 = (unsigned)(uintptr_t)(lds_f*)&cs[2][threadIdx.x];

  for (int t0 = 0; t0 < NPTS; t0 += 256) {
    __syncthreads();
    const float* g0 = x2 + t0 + threadIdx.x;
    const float* g1 = x2 + NPTS + t0 + threadIdx.x;
    const float* g2 = x2 + 2*NPTS + t0 + threadIdx.x;
    asm volatile("global_load_async_to_lds_b32 %0, %1, off" :: "v"(l0), "v"(g0) : "memory");
    asm volatile("global_load_async_to_lds_b32 %0, %1, off" :: "v"(l1), "v"(g1) : "memory");
    asm volatile("global_load_async_to_lds_b32 %0, %1, off" :: "v"(l2), "v"(g2) : "memory");
    asm volatile("s_wait_asynccnt 0" ::: "memory");
    __syncthreads();
    for (int j = 0; j < 256; ++j) {
      float dx = cs[0][j] - px, dy = cs[1][j] - py, dz = cs[2][j] - pz;
      float d = dx*dx + dy*dy + dz*dz;
      if (d < bd[KNN-1]) {
        float dv = d; int iv = t0 + j;
#pragma unroll
        for (int m = 0; m < KNN; ++m) {
          bool cnd = dv < bd[m];
          float td = bd[m]; int ti = bi[m];
          bd[m] = cnd ? dv : bd[m]; bi[m] = cnd ? iv : bi[m];
          dv = cnd ? td : dv;       iv = cnd ? ti : iv;
        }
      }
    }
  }
  int* o = idxb + ((size_t)b*NPTS + n) * KNN;
#pragma unroll
  for (int j = 0; j < KNN; ++j) o[j] = bi[j];
}

// ---------------------------------------------------------------------------
// base: fuse GEMMs (WMMA) + fold dc/df/te/bias terms into per-(b,n) vectors
// ---------------------------------------------------------------------------
__global__ __launch_bounds__(256) void base_kernel(
    const float* points1, const float* dfbuf,
    const _Float16* wfusef, const float* wcols, const float* tconst,
    const float* br0, const float* bz0, const float* bh0,
    float* base_r, float* base_z, float* base_h, float* g_ro)
{
  __shared__ _Float16 P1s[64 * HSTR];
  int b  = blockIdx.x / (NPTS/64);
  int n0 = (blockIdx.x % (NPTS/64)) * 64;
  int tid = threadIdx.x, wave = tid >> 5, lane = tid & 31;
  for (int i = tid; i < 64*HCH; i += 256) {
    int q = i & 63, ch = i >> 6;
    P1s[q*HSTR + ch] = (_Float16)points1[((size_t)b*HCH + ch)*NPTS + n0 + q];
  }
  __syncthreads();
  int mt = wave;
  int l16 = lane & 15, hi = (lane >> 4) << 3;
  int chb = mt*16 + hi;
  for (int br = 0; br < 3; ++br) {
    const _Float16* W = wfusef + (size_t)br*HCH*HCH;
    const float* bias = (br == 0) ? br0 : (br == 1) ? bz0 : bh0;
    float4 tc0 = *(const float4*)(tconst + (br*NB + b)*HCH + chb);
    float4 tc1 = *(const float4*)(tconst + (br*NB + b)*HCH + chb + 4);
    float4 bi0 = *(const float4*)(bias + chb);
    float4 bi1 = *(const float4*)(bias + chb + 4);
    v16h a[4];
#pragma unroll
    for (int ks = 0; ks < 4; ++ks) a[ks] = ldA(W, HCH, mt, ks, lane);
    for (int nt = 0; nt < 4; ++nt) {
      v8f acc = zero8();
#pragma unroll
      for (int ks = 0; ks < 4; ++ks)
        acc = wmma_f16(a[ks], ldB(P1s, HSTR, nt, ks, lane), acc);
      int q = nt*16 + l16;
      size_t nidx = (size_t)b*NPTS + n0 + q;
      float4 dfv = *(const float4*)(dfbuf + nidx*4);   // df0,df1,df2,dce
      float e[8];
#pragma unroll
      for (int j = 0; j < 8; ++j) {
        float4 w4 = *(const float4*)(wcols + (size_t)((br*HCH + chb + j) << 2));
        float tcv = (j < 4) ? ((const float*)&tc0)[j] : ((const float*)&tc1)[j-4];
        float bv  = (j < 4) ? ((const float*)&bi0)[j] : ((const float*)&bi1)[j-4];
        e[j] = w4.x*dfv.w + w4.y*dfv.x + w4.z*dfv.y + w4.w*dfv.z + tcv + bv;
      }
      float* outp = ((br == 0) ? base_r : (br == 1) ? base_z : base_h) + (nidx << 7) + chb;
      float4 o0, o1;
      if (br < 2) {
        o0 = make_float4(acc[0]+e[0], acc[1]+e[1], acc[2]+e[2], acc[3]+e[3]);
        o1 = make_float4(acc[4]+e[4], acc[5]+e[5], acc[6]+e[6], acc[7]+e[7]);
      } else {
        o0 = make_float4(e[0], e[1], e[2], e[3]);
        o1 = make_float4(e[4], e[5], e[6], e[7]);
        float* gp = g_ro + (nidx << 7) + chb;
        *(float4*)(gp)     = make_float4(acc[0], acc[1], acc[2], acc[3]);
        *(float4*)(gp + 4) = make_float4(acc[4], acc[5], acc[6], acc[7]);
      }
      *(float4*)(outp)     = o0;
      *(float4*)(outp + 4) = o1;
    }
  }
}

// ---------------------------------------------------------------------------
// gru: fused gather + 9 WMMA GEMM stages + pools + GRU update + head
// ---------------------------------------------------------------------------
__global__ __launch_bounds__(256) void gru_kernel(
    const float* __restrict__ xyz1, const float* __restrict__ xyz2,
    const float* __restrict__ points1, const _Float16* __restrict__ p2t,
    const int* __restrict__ idxb, const float* __restrict__ fnbuf,
    const float* __restrict__ base_r, const float* __restrict__ base_z,
    const float* __restrict__ base_h, const float* __restrict__ g_ro,
    const _Float16* __restrict__ w0f, const _Float16* __restrict__ w12f,
    const float* __restrict__ br12, const float* __restrict__ bz12,
    const float* __restrict__ bh12,
    const float* __restrict__ w_fc, const float* __restrict__ b_fc,
    float* __restrict__ out)
{
  extern __shared__ char smem[];
  _Float16* Xs   = (_Float16*)smem;               // [NPOS][XSTR]
  _Float16* BufA = Xs   + NPOS*XSTR;              // [NPOS][HSTR]
  _Float16* BufB = BufA + NPOS*HSTR;
  _Float16* BufR = BufB + NPOS*HSTR;
  _Float16* Ps   = BufR + NPOS*HSTR;              // [16][HSTR]
  float* zout    = (float*)(Ps + 16*HSTR);        // [HCH][NQ]
  float* hout    = zout + HCH*NQ;

  int b  = blockIdx.x / (NPTS/NQ);
  int n0 = (blockIdx.x % (NPTS/NQ)) * NQ;
  int tid = threadIdx.x, wave = tid >> 5, lane = tid & 31;
  int mt = wave;
  int l16 = lane & 15, hi = (lane >> 4) << 3;
  int chb = mt*16 + hi;

  const _Float16* Wr0 = w0f + 0*HCH*CINP;
  const _Float16* Wz0 = w0f + 1*HCH*CINP;
  const _Float16* Wh0 = w0f + 2*HCH*CINP;
  const _Float16* Wr1 = w12f + 0*HCH*HCH;
  const _Float16* Wr2 = w12f + 1*HCH*HCH;
  const _Float16* Wz1 = w12f + 2*HCH*HCH;
  const _Float16* Wz2 = w12f + 3*HCH*HCH;
  const _Float16* Wh1 = w12f + 4*HCH*HCH;
  const _Float16* Wh2 = w12f + 5*HCH*HCH;

  // ---- Stage 0: vectorized gather into LDS (2 threads per position) ----
  {
    int pos = tid >> 1, hsel = tid & 1;
    int q = pos >> 4, k = pos & 15;
    int n = n0 + q;
    int nb = idxb[((size_t)b*NPTS + n)*KNN + k];
    const v8h* src = (const v8h*)(p2t + ((size_t)b*NPTS + nb) * 64);
    v8h* dst = (v8h*)(Xs + pos*XSTR);
#pragma unroll
    for (int c2 = 0; c2 < 4; ++c2) dst[hsel*4 + c2] = src[hsel*4 + c2];
    _Float16* xr = Xs + pos*XSTR;
    if (hsel == 0) {
#pragma unroll
      for (int c = 0; c < 3; ++c)
        xr[64+c] = (_Float16)(xyz2[((size_t)b*3+c)*NPTS + nb] - xyz1[((size_t)b*3+c)*NPTS + n]);
    } else {
      for (int c = 67; c < XSTR; ++c) xr[c] = (_Float16)0.0f;
    }
    __builtin_prefetch(w0f  + (size_t)tid*144, 0, 3);   // warm weight stream
    __builtin_prefetch(w12f + (size_t)tid*384, 0, 3);
  }
  __syncthreads();

  // ---- r branch ----
  gemm8<3>(Wr0, CINP, Xs, XSTR, mt, lane, [&](int nt, v8f acc) {
    int nloc = nt*16 + l16;
    const float* bs = base_r + (((size_t)b*NPTS + n0 + nt) << 7) + chb;
    float4 b0 = *(const float4*)(bs);
    float4 b1 = *(const float4*)(bs + 4);
    v8h o;
    o[0] = (_Float16)leaky(acc[0] + b0.x); o[1] = (_Float16)leaky(acc[1] + b0.y);
    o[2] = (_Float16)leaky(acc[2] + b0.z); o[3] = (_Float16)leaky(acc[3] + b0.w);
    o[4] = (_Float16)leaky(acc[4] + b1.x); o[5] = (_Float16)leaky(acc[5] + b1.y);
    o[6] = (_Float16)leaky(acc[6] + b1.z); o[7] = (_Float16)leaky(acc[7] + b1.w);
    *(v8h*)(BufA + nloc*HSTR + chb) = o;
  });
  __syncthreads();
  {
    float4 c0 = *(const float4*)(br12 + chb);
    float4 c1 = *(const float4*)(br12 + chb + 4);
    gemm8<4>(Wr1, HCH, BufA, HSTR, mt, lane, [&](int nt, v8f acc) {
      int nloc = nt*16 + l16;
      v8h o;
      o[0] = (_Float16)leaky(acc[0] + c0.x); o[1] = (_Float16)leaky(acc[1] + c0.y);
      o[2] = (_Float16)leaky(acc[2] + c0.z); o[3] = (_Float16)leaky(acc[3] + c0.w);
      o[4] = (_Float16)leaky(acc[4] + c1.x); o[5] = (_Float16)leaky(acc[5] + c1.y);
      o[6] = (_Float16)leaky(acc[6] + c1.z); o[7] = (_Float16)leaky(acc[7] + c1.w);
      *(v8h*)(BufB + nloc*HSTR + chb) = o;
    });
  }
  __syncthreads();
  {
    float4 c0 = *(const float4*)(br12 + HCH + chb);
    float4 c1 = *(const float4*)(br12 + HCH + chb + 4);
    gemm8<4>(Wr2, HCH, BufB, HSTR, mt, lane, [&](int nt, v8f acc) {
      int nloc = nt*16 + l16;
      v8h o;
      o[0] = (_Float16)sigm(acc[0] + c0.x); o[1] = (_Float16)sigm(acc[1] + c0.y);
      o[2] = (_Float16)sigm(acc[2] + c0.z); o[3] = (_Float16)sigm(acc[3] + c0.w);
      o[4] = (_Float16)sigm(acc[4] + c1.x); o[5] = (_Float16)sigm(acc[5] + c1.y);
      o[6] = (_Float16)sigm(acc[6] + c1.z); o[7] = (_Float16)sigm(acc[7] + c1.w);
      *(v8h*)(BufR + nloc*HSTR + chb) = o;
    });
  }
  __syncthreads();

  // ---- z branch ----
  gemm8<3>(Wz0, CINP, Xs, XSTR, mt, lane, [&](int nt, v8f acc) {
    int nloc = nt*16 + l16;
    const float* bs = base_z + (((size_t)b*NPTS + n0 + nt) << 7) + chb;
    float4 b0 = *(const float4*)(bs);
    float4 b1 = *(const float4*)(bs + 4);
    v8h o;
    o[0] = (_Float16)leaky(acc[0] + b0.x); o[1] = (_Float16)leaky(acc[1] + b0.y);
    o[2] = (_Float16)leaky(acc[2] + b0.z); o[3] = (_Float16)leaky(acc[3] + b0.w);
    o[4] = (_Float16)leaky(acc[4] + b1.x); o[5] = (_Float16)leaky(acc[5] + b1.y);
    o[6] = (_Float16)leaky(acc[6] + b1.z); o[7] = (_Float16)leaky(acc[7] + b1.w);
    *(v8h*)(BufA + nloc*HSTR + chb) = o;
  });
  __syncthreads();
  {
    float4 c0 = *(const float4*)(bz12 + chb);
    float4 c1 = *(const float4*)(bz12 + chb + 4);
    gemm8<4>(Wz1, HCH, BufA, HSTR, mt, lane, [&](int nt, v8f acc) {
      int nloc = nt*16 + l16;
      v8h o;
      o[0] = (_Float16)leaky(acc[0] + c0.x); o[1] = (_Float16)leaky(acc[1] + c0.y);
      o[2] = (_Float16)leaky(acc[2] + c0.z); o[3] = (_Float16)leaky(acc[3] + c0.w);
      o[4] = (_Float16)leaky(acc[4] + c1.x); o[5] = (_Float16)leaky(acc[5] + c1.y);
      o[6] = (_Float16)leaky(acc[6] + c1.z); o[7] = (_Float16)leaky(acc[7] + c1.w);
      *(v8h*)(BufB + nloc*HSTR + chb) = o;
    });
  }
  __syncthreads();
  for (int i = tid; i < 16*HCH; i += 256) {   // max-pool over K, zero tail rows
    int q = i >> 7, ch = i & 127;
    float m = 0.f;
    if (q < NQ) {
      m = -3.0e38f;
#pragma unroll
      for (int k = 0; k < KNN; ++k) m = fmaxf(m, (float)BufB[(q*16 + k)*HSTR + ch]);
    }
    Ps[q*HSTR + ch] = (_Float16)m;
  }
  __syncthreads();
  gemm1<4>(Wz2, HCH, Ps, HSTR, mt, lane, [&](v8f acc) {
    int q = l16;
    if (q < NQ) {
#pragma unroll
      for (int j = 0; j < 8; ++j)
        zout[(chb + j)*NQ + q] = sigm(acc[j] + bz12[HCH + chb + j]);
    }
  });
  __syncthreads();

  // ---- h branch (uses g * r at first layer) ----
  gemm8<3>(Wh0, CINP, Xs, XSTR, mt, lane, [&](int nt, v8f acc) {
    int nloc = nt*16 + l16;
    size_t base = (((size_t)b*NPTS + n0 + nt) << 7) + chb;
    float4 b0 = *(const float4*)(base_h + base);
    float4 b1 = *(const float4*)(base_h + base + 4);
    float4 g0 = *(const float4*)(g_ro + base);
    float4 g1 = *(const float4*)(g_ro + base + 4);
    v8h rv = *(const v8h*)(BufR + nloc*HSTR + chb);
    v8h o;
    o[0] = (_Float16)leaky(acc[0] + b0.x + g0.x*(float)rv[0]);
    o[1] = (_Float16)leaky(acc[1] + b0.y + g0.y*(float)rv[1]);
    o[2] = (_Float16)leaky(acc[2] + b0.z + g0.z*(float)rv[2]);
    o[3] = (_Float16)leaky(acc[3] + b0.w + g0.w*(float)rv[3]);
    o[4] = (_Float16)leaky(acc[4] + b1.x + g1.x*(float)rv[4]);
    o[5] = (_Float16)leaky(acc[5] + b1.y + g1.y*(float)rv[5]);
    o[6] = (_Float16)leaky(acc[6] + b1.z + g1.z*(float)rv[6]);
    o[7] = (_Float16)leaky(acc[7] + b1.w + g1.w*(float)rv[7]);
    *(v8h*)(BufA + nloc*HSTR + chb) = o;
  });
  __syncthreads();
  {
    float4 c0 = *(const float4*)(bh12 + chb);
    float4 c1 = *(const float4*)(bh12 + chb + 4);
    gemm8<4>(Wh1, HCH, BufA, HSTR, mt, lane, [&](int nt, v8f acc) {
      int nloc = nt*16 + l16;
      v8h o;
      o[0] = (_Float16)leaky(acc[0] + c0.x); o[1] = (_Float16)leaky(acc[1] + c0.y);
      o[2] = (_Float16)leaky(acc[2] + c0.z); o[3] = (_Float16)leaky(acc[3] + c0.w);
      o[4] = (_Float16)leaky(acc[4] + c1.x); o[5] = (_Float16)leaky(acc[5] + c1.y);
      o[6] = (_Float16)leaky(acc[6] + c1.z); o[7] = (_Float16)leaky(acc[7] + c1.w);
      *(v8h*)(BufB + nloc*HSTR + chb) = o;
    });
  }
  __syncthreads();
  for (int i = tid; i < 16*HCH; i += 256) {
    int q = i >> 7, ch = i & 127;
    float m = 0.f;
    if (q < NQ) {
      m = -3.0e38f;
#pragma unroll
      for (int k = 0; k < KNN; ++k) m = fmaxf(m, (float)BufB[(q*16 + k)*HSTR + ch]);
    }
    Ps[q*HSTR + ch] = (_Float16)m;
  }
  __syncthreads();
  gemm1<4>(Wh2, HCH, Ps, HSTR, mt, lane, [&](v8f acc) {
    int q = l16;
    if (q < NQ) {
#pragma unroll
      for (int j = 0; j < 8; ++j)
        hout[(chb + j)*NQ + q] = tanhf(acc[j] + bh12[HCH + chb + j]);
    }
  });
  __syncthreads();

  // ---- GRU update + 128->4 head + residual output ----
  {
    int q = wave;                 // 8 waves, one query each
    int n = n0 + q;
    float a0 = 0.f, a1 = 0.f, a2 = 0.f, a3 = 0.f;
    for (int ch = lane; ch < HCH; ch += 32) {
      float z  = zout[ch*NQ + q];
      float h  = hout[ch*NQ + q];
      float p1 = points1[((size_t)b*HCH + ch)*NPTS + n];
      float np = (1.f - z)*p1 + z*h;
      a0 += w_fc[ch]        * np;
      a1 += w_fc[HCH  + ch] * np;
      a2 += w_fc[2*HCH+ ch] * np;
      a3 += w_fc[3*HCH+ ch] * np;
    }
#pragma unroll
    for (int off = 16; off > 0; off >>= 1) {
      a0 += __shfl_down(a0, off);
      a1 += __shfl_down(a1, off);
      a2 += __shfl_down(a2, off);
      a3 += __shfl_down(a3, off);
    }
    if (lane == 0) {
      const float* fn = fnbuf + ((size_t)b*NPTS + n)*4;
      float u0 = fminf(fmaxf(a0 + b_fc[0], -200.f), 200.f);
      float u1 = fminf(fmaxf(a1 + b_fc[1], -200.f), 200.f);
      float u2 = fminf(fmaxf(a2 + b_fc[2], -200.f), 200.f);
      float u3 = a3 + b_fc[3];
      out[((size_t)b*4 + 0)*NPTS + n] = fn[0] + u0;
      out[((size_t)b*4 + 1)*NPTS + n] = fn[1] + u1;
      out[((size_t)b*4 + 2)*NPTS + n] = fn[2] + u2;
      out[((size_t)b*4 + 3)*NPTS + n] = fn[3] + u3;
    }
  }
}

// ---------------------------------------------------------------------------
extern "C" void kernel_launch(void* const* d_in, const int* in_sizes, int n_in,
                              void* d_out, int out_size, void* d_ws, size_t ws_size,
                              hipStream_t stream) {
  (void)in_sizes; (void)n_in; (void)out_size; (void)ws_size;
  const float* xyz1       = (const float*)d_in[0];
  const float* xyz2       = (const float*)d_in[1];
  const float* points1    = (const float*)d_in[2];
  const float* points2    = (const float*)d_in[3];
  const float* flow       = (const float*)d_in[4];
  const float* flow_gt    = (const float*)d_in[5];
  const float* certainty  = (const float*)d_in[6];
  const float* noise      = (const float*)d_in[7];
  const float* noise_cert = (const float*)d_in[8];
  const int*   t_in       = (const int*)d_in[9];
  const float* w_time1    = (const float*)d_in[10];
  const float* b_time1    = (const float*)d_in[11];
  const float* w_time2    = (const float*)d_in[12];
  const float* b_time2    = (const float*)d_in[13];
  const float* w_fuse_r   = (const float*)d_in[14];
  const float* w_fuse_z   = (const float*)d_in[15];
  const float* w_fuse_ro  = (const float*)d_in[16];
  const float* wr0        = (const float*)d_in[17];
  const float* br0        = (const float*)d_in[18];
  const float* wr12       = (const float*)d_in[19];
  const float* br12       = (const float*)d_in[20];
  const float* wz0        = (const float*)d_in[21];
  const float* bz0        = (const float*)d_in[22];
  const float* wz12       = (const float*)d_in[23];
  const float* bz12       = (const float*)d_in[24];
  const float* wh0        = (const float*)d_in[25];
  const float* bh0        = (const float*)d_in[26];
  const float* wh12       = (const float*)d_in[27];
  const float* bh12       = (const float*)d_in[28];
  const float* w_fc       = (const float*)d_in[29];
  const float* b_fc       = (const float*)d_in[30];

  char* wsb = (char*)d_ws;
  size_t off = 0;
  auto carve = [&](size_t bytes) -> char* {
    char* p = wsb + off;
    off += (bytes + 255) & ~(size_t)255;
    return p;
  };
  float* sched   = (float*)carve(4 * sizeof(float));
  float* te      = (float*)carve(NB*64 * sizeof(float));
  float* tconst  = (float*)carve(3*NB*HCH * sizeof(float));
  float* wcols   = (float*)carve(3*HCH*4 * sizeof(float));
  float* dfbuf   = (float*)carve((size_t)NB*NPTS*4 * sizeof(float));
  float* fnbuf   = (float*)carve((size_t)NB*NPTS*4 * sizeof(float));
  float* base_r  = (float*)carve((size_t)NB*NPTS*HCH * sizeof(float));
  float* base_z  = (float*)carve((size_t)NB*NPTS*HCH * sizeof(float));
  float* base_h  = (float*)carve((size_t)NB*NPTS*HCH * sizeof(float));
  float* g_ro    = (float*)carve((size_t)NB*NPTS*HCH * sizeof(float));
  int*   idxb    = (int*)carve((size_t)NB*NPTS*KNN * sizeof(int));
  _Float16* p2t    = (_Float16*)carve((size_t)NB*NPTS*64 * 2);
  _Float16* w0f    = (_Float16*)carve((size_t)3*HCH*CINP * 2);
  _Float16* w12f   = (_Float16*)carve((size_t)3*2*HCH*HCH * 2);
  _Float16* wfusef = (_Float16*)carve((size_t)3*HCH*HCH * 2);

  convert_kernel<<<96, 256, 0, stream>>>(wr0, wz0, wh0, wr12, wz12, wh12,
                                         w_fuse_r, w_fuse_z, w_fuse_ro,
                                         w0f, w12f, wfusef, wcols);
  p2half_kernel<<<(NB*NPTS*64)/256, 256, 0, stream>>>(points2, p2t);
  prep0_kernel<<<1, 256, 0, stream>>>(t_in, w_time1, b_time1, w_time2, b_time2,
                                      wr0, wz0, wh0, sched, te, tconst);
  prep1_kernel<<<NB*NPTS/256, 256, 0, stream>>>(flow, flow_gt, certainty, noise,
                                                noise_cert, sched, dfbuf, fnbuf);
  knn_kernel<<<NB*(NPTS/256), 256, 0, stream>>>(xyz1, xyz2, idxb);
  base_kernel<<<NB*(NPTS/64), 256, 0, stream>>>(points1, dfbuf, wfusef, wcols, tconst,
                                                br0, bz0, bh0,
                                                base_r, base_z, base_h, g_ro);
  size_t smem = (size_t)(NPOS*XSTR + 3*NPOS*HSTR + 16*HSTR) * sizeof(_Float16)
              + (size_t)2*HCH*NQ * sizeof(float);
  hipFuncSetAttribute(reinterpret_cast<const void*>(gru_kernel),
                      hipFuncAttributeMaxDynamicSharedMemorySize, (int)smem);
  gru_kernel<<<NB*(NPTS/NQ), 256, smem, stream>>>(
      xyz1, xyz2, points1, p2t, idxb, fnbuf,
      base_r, base_z, base_h, g_ro, w0f, w12f,
      br12, bz12, bh12, w_fc, b_fc, (float*)d_out);
}
